// ScoreNet_1065151889700
// MI455X (gfx1250) — matrix operand order repaired and checked by reading, not compiled, not run on hardware
//
#include <hip/hip_runtime.h>
#include <math.h>

// ---------------- types ----------------
typedef __attribute__((ext_vector_type(16))) __bf16 bf16x16;
typedef __attribute__((ext_vector_type(8)))  __bf16 bf16x8;
typedef __attribute__((ext_vector_type(8)))  float  v8f;

#define H 256
#define WAVES 8   // waves per block in GEMM kernels (8*16 = 128 rows per block)

// ---------------- WMMA fragment helpers ----------------
// B packed layout: [kt][nt][lane 0..31][i 0..15] bf16
//   lane = (half ? 16 : 0) + (n & 15) ; i<8 -> K = kt*32 + half*8 + i ; i>=8 -> K = kt*32 + 16 + half*8 + (i-8) ; N = nt*16 + (lane&15)
__device__ __forceinline__ bf16x16 load_b_frag(const __bf16* Bp, int kt, int nt,
                                               int nTiles, int lane) {
    const __bf16* p = Bp + (((size_t)kt * nTiles + nt) * 32 + lane) * 16;
    return *(const bf16x16*)p;  // 32B contiguous per lane
}

// A fragment from a row-major bf16 matrix [M x K]:
//   lane: M = lane&15 ; i<8 -> K = kt*32 + half*8 + i ; i>=8 -> K = kt*32 + 16 + half*8 + (i-8)
__device__ __forceinline__ bf16x16 load_a_bf16(const __bf16* A, size_t row, int K,
                                               int kt, int half) {
    const __bf16* p = A + row * (size_t)K + kt * 32 + half * 8;
    bf16x8 lo = *(const bf16x8*)p;
    bf16x8 hi = *(const bf16x8*)(p + 16);
    bf16x16 r;
#pragma unroll
    for (int i = 0; i < 8; ++i) { r[i] = lo[i]; r[8 + i] = hi[i]; }
    return r;
}

// ---------------- weight packing (fp32 -> bf16 WMMA-B layout) ----------------
__global__ void pack_b_kernel(const float* __restrict__ B, __bf16* __restrict__ Bp,
                              int K, int N) {
    int idx = blockIdx.x * blockDim.x + threadIdx.x;
    if (idx >= K * N) return;
    int k = idx / N, n = idx % N;
    int kt = k >> 5, kin = k & 31;
    int lane = ((kin & 8) << 1) + (n & 15);
    int i = (kin & 7) | ((kin >> 1) & 8);
    int nt = n >> 4;
    int nTiles = N >> 4;
    Bp[(((size_t)kt * nTiles + nt) * 32 + lane) * 16 + i] = (__bf16)B[idx];
}

// ---------------- time embedding: per-graph scalar temb + 1/std ----------------
__global__ void temb_kernel(const float* __restrict__ t, const float* __restrict__ fw,
                            const float* __restrict__ tW, const float* __restrict__ tb,
                            const float* __restrict__ d1W, const float* __restrict__ d1b,
                            float* __restrict__ temb, float* __restrict__ stdinv) {
    __shared__ float f[H];
    __shared__ float red[H];
    int g = blockIdx.x, j = threadIdx.x;
    float tv = t[g];
    if (j < H / 2) {
        float xp = tv * fw[j] * 6.28318530717958647692f;
        f[j] = sinf(xp);
        f[j + H / 2] = cosf(xp);
    }
    __syncthreads();
    float y = tb[j];
    for (int k = 0; k < H; ++k) y += f[k] * tW[k * H + j];
    red[j] = y * d1W[j];
    __syncthreads();
    for (int s = H / 2; s > 0; s >>= 1) {
        if (j < s) red[j] += red[j + s];
        __syncthreads();
    }
    if (j == 0) {
        temb[g] = red[0] + d1b[0];
        const float ls = 3.21887582486820075f;  // ln(25)
        float var = (expf(2.f * tv * ls) - 1.f) / (2.f * ls);
        stdinv[g] = rsqrtf(var);
    }
}

// ---------------- per-edge graph lookups ----------------
__global__ void e2g_kernel(const int* __restrict__ src, const int* __restrict__ batch,
                           const float* __restrict__ temb, const float* __restrict__ stdinv,
                           float* __restrict__ tembE, float* __restrict__ stdinvE, int E) {
    int e = blockIdx.x * blockDim.x + threadIdx.x;
    if (e >= E) return;
    int g = batch[src[e]];
    tembE[e] = temb[g];
    stdinvE[e] = stdinv[g];
}

// ---------------- atom embedding gather -> h0 ----------------
__global__ void gather_atoms_kernel(const int* __restrict__ ntype,
                                    const float* __restrict__ emb,
                                    float* __restrict__ h, int N) {
    long tid = (long)blockIdx.x * blockDim.x + threadIdx.x;
    int n = (int)(tid >> 6);
    if (n >= N) return;
    int c = (int)(tid & 63) << 2;
    float4 v = *(const float4*)(emb + (size_t)ntype[n] * H + c);
    *(float4*)(h + (size_t)n * H + c) = v;
}

__global__ void zero_kernel(float* __restrict__ p, long n4) {
    long tid = (long)blockIdx.x * blockDim.x + threadIdx.x;
    if (tid < n4) ((float4*)p)[tid] = make_float4(0.f, 0.f, 0.f, 0.f);
}

// ---------------- GEMM: bond_attr = (relu(len*W1+b1) @ W2 + b2 + temb) * bond_emb ----------------
__global__ void dembed_gemm_kernel(const float* __restrict__ elen,
                                   const float* __restrict__ W1, const float* __restrict__ b1,
                                   const __bf16* __restrict__ Bp, const float* __restrict__ b2,
                                   const float* __restrict__ tembE, const int* __restrict__ etype,
                                   const float* __restrict__ bond_emb,
                                   __bf16* __restrict__ bond, int M) {
    int lane = threadIdx.x & 31, wv = threadIdx.x >> 5;
    int m0 = (blockIdx.x * WAVES + wv) << 4;
    int n0 = blockIdx.y << 4;
    int hf = lane >> 4;
    int mrow = m0 + (lane & 15);
    int arow = mrow < M ? mrow : M - 1;
    float len = elen[arow];
    v8f acc = {};
    for (int kt = 0; kt < H / 32; ++kt) {
        int kb = kt * 32 + hf * 8;
        bf16x16 a;
#pragma unroll
        for (int i = 0; i < 8; ++i) {
            a[i]     = (__bf16)fmaxf(len * W1[kb + i]      + b1[kb + i],      0.f);
            a[8 + i] = (__bf16)fmaxf(len * W1[kb + 16 + i] + b1[kb + 16 + i], 0.f);
        }
        bf16x16 b = load_b_frag(Bp, kt, blockIdx.y, H / 16, lane);
        acc = __builtin_amdgcn_wmma_f32_16x16x32_bf16(false, a, false, b, (short)0, acc,
                                                      false, false);
    }
    int col = n0 + (lane & 15);
    float bias = b2[col];
#pragma unroll
    for (int r = 0; r < 8; ++r) {
        int row = m0 + r + hf * 8;
        if (row < M) {
            float bm = bond_emb[(size_t)etype[row] * H + col];
            bond[(size_t)row * H + col] = (__bf16)((acc[r] + bias + tembE[row]) * bm);
        }
    }
}

// ---------------- message + scatter-add (one wave per edge) ----------------
__global__ void message_kernel(const float* __restrict__ h, const __bf16* __restrict__ bond,
                               const int* __restrict__ src, const int* __restrict__ dst,
                               float* __restrict__ aggr, int E) {
    long tid = (long)blockIdx.x * blockDim.x + threadIdx.x;
    int e = (int)(tid >> 5);
    if (e >= E) return;
    int c = (int)(tid & 31) << 3;
    int s = src[e], d = dst[e];
    const float* ph = h + (size_t)s * H + c;
    const __bf16* pb = bond + (size_t)e * H + c;
    float* pa = aggr + (size_t)d * H + c;
#pragma unroll
    for (int i = 0; i < 8; ++i) {
        float m = fmaxf(ph[i] + (float)pb[i], 0.f);
        atomicAdd(pa + i, m);
    }
}

// ---------------- GIN GEMM1: tmp = relu((h + aggr) @ W1 + b1) (bf16 out) ----------------
__global__ void gin_gemm1_kernel(const float* __restrict__ h, const float* __restrict__ aggr,
                                 const __bf16* __restrict__ Bp, const float* __restrict__ b1,
                                 __bf16* __restrict__ tmp, int M) {
    int lane = threadIdx.x & 31, wv = threadIdx.x >> 5;
    int m0 = (blockIdx.x * WAVES + wv) << 4;
    int n0 = blockIdx.y << 4;
    int hf = lane >> 4;
    int mrow = m0 + (lane & 15);
    size_t arow = (size_t)(mrow < M ? mrow : M - 1);
    v8f acc = {};
    for (int kt = 0; kt < H / 32; ++kt) {
        int kb = kt * 32 + hf * 8;
        const float* ph = h + arow * H + kb;
        const float* pa = aggr + arow * H + kb;
        bf16x16 a;
#pragma unroll
        for (int i = 0; i < 8; ++i) {
            a[i]     = (__bf16)(ph[i]      + pa[i]);
            a[8 + i] = (__bf16)(ph[16 + i] + pa[16 + i]);
        }
        bf16x16 b = load_b_frag(Bp, kt, blockIdx.y, H / 16, lane);
        acc = __builtin_amdgcn_wmma_f32_16x16x32_bf16(false, a, false, b, (short)0, acc,
                                                      false, false);
    }
    int col = n0 + (lane & 15);
    float bias = b1[col];
#pragma unroll
    for (int r = 0; r < 8; ++r) {
        int row = m0 + r + hf * 8;
        if (row < M)
            tmp[(size_t)row * H + col] = (__bf16)fmaxf(acc[r] + bias, 0.f);
    }
}

// ---------------- GIN GEMM2: h_out = act(tmp @ W2 + b2) + h_in ----------------
__global__ void gin_gemm2_kernel(const __bf16* __restrict__ tmp, const __bf16* __restrict__ Bp,
                                 const float* __restrict__ b2, const float* __restrict__ h_in,
                                 float* __restrict__ h_out, int M, int doRelu) {
    int lane = threadIdx.x & 31, wv = threadIdx.x >> 5;
    int m0 = (blockIdx.x * WAVES + wv) << 4;
    int n0 = blockIdx.y << 4;
    int hf = lane >> 4;
    int mrow = m0 + (lane & 15);
    size_t arow = (size_t)(mrow < M ? mrow : M - 1);
    v8f acc = {};
    for (int kt = 0; kt < H / 32; ++kt) {
        bf16x16 a = load_a_bf16(tmp, arow, H, kt, hf);
        bf16x16 b = load_b_frag(Bp, kt, blockIdx.y, H / 16, lane);
        acc = __builtin_amdgcn_wmma_f32_16x16x32_bf16(false, a, false, b, (short)0, acc,
                                                      false, false);
    }
    int col = n0 + (lane & 15);
    float bias = b2[col];
#pragma unroll
    for (int r = 0; r < 8; ++r) {
        int row = m0 + r + hf * 8;
        if (row < M) {
            float v = acc[r] + bias;
            if (doRelu) v = fmaxf(v, 0.f);
            h_out[(size_t)row * H + col] = v + h_in[(size_t)row * H + col];
        }
    }
}

// ---------------- OUT GEMM1: y1 = relu([h_src*h_dst, bond] @ out_W1 + b1), K=512 ----------------
__global__ void out_gemm1_kernel(const float* __restrict__ h, const __bf16* __restrict__ bond,
                                 const int* __restrict__ src, const int* __restrict__ dst,
                                 const __bf16* __restrict__ Bp, const float* __restrict__ b1,
                                 __bf16* __restrict__ y1, int M) {
    int lane = threadIdx.x & 31, wv = threadIdx.x >> 5;
    int m0 = (blockIdx.x * WAVES + wv) << 4;
    int n0 = blockIdx.y << 4;
    int hf = lane >> 4;
    int mrow = m0 + (lane & 15);
    size_t arow = (size_t)(mrow < M ? mrow : M - 1);
    size_t sb = (size_t)src[arow] * H;
    size_t db = (size_t)dst[arow] * H;
    v8f acc = {};
    for (int kt = 0; kt < 16; ++kt) {  // K = 512
        bf16x16 a;
        if (kt < 8) {
            int kb = kt * 32 + hf * 8;
            const float* ps = h + sb + kb;
            const float* pd = h + db + kb;
#pragma unroll
            for (int i = 0; i < 8; ++i) {
                a[i]     = (__bf16)(ps[i]      * pd[i]);
                a[8 + i] = (__bf16)(ps[16 + i] * pd[16 + i]);
            }
        } else {
            a = load_a_bf16(bond, arow, H, kt - 8, hf);
        }
        bf16x16 b = load_b_frag(Bp, kt, blockIdx.y, H / 16, lane);
        acc = __builtin_amdgcn_wmma_f32_16x16x32_bf16(false, a, false, b, (short)0, acc,
                                                      false, false);
    }
    int col = n0 + (lane & 15);
    float bias = b1[col];
#pragma unroll
    for (int r = 0; r < 8; ++r) {
        int row = m0 + r + hf * 8;
        if (row < M)
            y1[(size_t)row * H + col] = (__bf16)fmaxf(acc[r] + bias, 0.f);
    }
}

// ---------------- OUT GEMM2: y2 = relu(y1 @ out_W2 + b2), N=128 ----------------
__global__ void out_gemm2_kernel(const __bf16* __restrict__ y1, const __bf16* __restrict__ Bp,
                                 const float* __restrict__ b2, __bf16* __restrict__ y2, int M) {
    int lane = threadIdx.x & 31, wv = threadIdx.x >> 5;
    int m0 = (blockIdx.x * WAVES + wv) << 4;
    int n0 = blockIdx.y << 4;
    int hf = lane >> 4;
    int mrow = m0 + (lane & 15);
    size_t arow = (size_t)(mrow < M ? mrow : M - 1);
    v8f acc = {};
    for (int kt = 0; kt < H / 32; ++kt) {
        bf16x16 a = load_a_bf16(y1, arow, H, kt, hf);
        bf16x16 b = load_b_frag(Bp, kt, blockIdx.y, 128 / 16, lane);
        acc = __builtin_amdgcn_wmma_f32_16x16x32_bf16(false, a, false, b, (short)0, acc,
                                                      false, false);
    }
    int col = n0 + (lane & 15);
    float bias = b2[col];
#pragma unroll
    for (int r = 0; r < 8; ++r) {
        int row = m0 + r + hf * 8;
        if (row < M)
            y2[(size_t)row * 128 + col] = (__bf16)fmaxf(acc[r] + bias, 0.f);
    }
}

// ---------------- final: scores = (y2 . out_W3 + b3) * stdinv  (wave per edge) ----------------
__global__ void final_kernel(const __bf16* __restrict__ y2, const float* __restrict__ w3,
                             const float* __restrict__ b3, const float* __restrict__ stdinvE,
                             float* __restrict__ out, int E) {
    int lane = threadIdx.x & 31;
    int e = blockIdx.x * (blockDim.x >> 5) + (threadIdx.x >> 5);
    if (e >= E) return;
    const __bf16* p = y2 + (size_t)e * 128 + lane * 4;
    float s = 0.f;
#pragma unroll
    for (int i = 0; i < 4; ++i) s += (float)p[i] * w3[lane * 4 + i];
#pragma unroll
    for (int off = 16; off > 0; off >>= 1) s += __shfl_xor(s, off, 32);
    if (lane == 0) out[e] = (s + b3[0]) * stdinvE[e];
}

// ---------------- host side ----------------
extern "C" void kernel_launch(void* const* d_in, const int* in_sizes, int n_in,
                              void* d_out, int out_size, void* d_ws, size_t ws_size,
                              hipStream_t stream) {
    const int N = in_sizes[0];          // 50000
    const int E = in_sizes[1];          // 300000
    const int G = in_sizes[5];          // 1024

    const int*   node_type = (const int*)d_in[0];
    const int*   edge_type = (const int*)d_in[1];
    const int*   src       = (const int*)d_in[2];
    const int*   dst       = ((const int*)d_in[2]) + E;
    const int*   batch     = (const int*)d_in[3];
    const float* elen      = (const float*)d_in[4];
    const float* t         = (const float*)d_in[5];
    const float* atom_emb  = (const float*)d_in[6];
    const float* bond_emb  = (const float*)d_in[7];
    const float* in_W1     = (const float*)d_in[8];
    const float* in_b1     = (const float*)d_in[9];
    const float* in_W2     = (const float*)d_in[10];
    const float* in_b2     = (const float*)d_in[11];
    const float* fw        = (const float*)d_in[12];
    const float* temb_W    = (const float*)d_in[13];
    const float* temb_b    = (const float*)d_in[14];
    const float* d1W       = (const float*)d_in[15];
    const float* d1b       = (const float*)d_in[16];
    const float* gin_W1    = (const float*)d_in[17];
    const float* gin_b1    = (const float*)d_in[18];
    const float* gin_W2    = (const float*)d_in[19];
    const float* gin_b2    = (const float*)d_in[20];
    const float* out_W1    = (const float*)d_in[21];
    const float* out_b1    = (const float*)d_in[22];
    const float* out_W2    = (const float*)d_in[23];
    const float* out_b2    = (const float*)d_in[24];
    const float* out_W3    = (const float*)d_in[25];
    const float* out_b3    = (const float*)d_in[26];
    float* scores = (float*)d_out;

    // workspace carve-out (256B aligned)
    char* ws = (char*)d_ws;
    size_t off = 0;
    auto carve = [&](size_t bytes) -> char* {
        char* p = ws + off;
        off += (bytes + 255) & ~(size_t)255;
        return p;
    };
    __bf16* bpInW2  = (__bf16*)carve((size_t)H * H * 2);
    __bf16* bpGin1  = (__bf16*)carve((size_t)4 * H * H * 2);
    __bf16* bpGin2  = (__bf16*)carve((size_t)4 * H * H * 2);
    __bf16* bpOutW1 = (__bf16*)carve((size_t)2 * H * H * 2);
    __bf16* bpOutW2 = (__bf16*)carve((size_t)H * 128 * 2);
    float*  tembG   = (float*)carve((size_t)G * 4);
    float*  stdinvG = (float*)carve((size_t)G * 4);
    float*  tembE   = (float*)carve((size_t)E * 4);
    float*  stdinvE = (float*)carve((size_t)E * 4);
    __bf16* bond    = (__bf16*)carve((size_t)E * H * 2);
    float*  hA      = (float*)carve((size_t)N * H * 4);
    float*  hB      = (float*)carve((size_t)N * H * 4);
    float*  aggr    = (float*)carve((size_t)N * H * 4);
    __bf16* tmp     = (__bf16*)carve((size_t)N * H * 2);
    __bf16* y1      = (__bf16*)carve((size_t)E * H * 2);
    __bf16* y2      = (__bf16*)carve((size_t)E * 128 * 2);

    // 1. pack all GEMM weights to bf16 WMMA-B layout
    {
        int thr = 256;
        int n1 = H * H;
        pack_b_kernel<<<(n1 + thr - 1) / thr, thr, 0, stream>>>(in_W2, bpInW2, H, H);
        for (int c = 0; c < 4; ++c) {
            pack_b_kernel<<<(n1 + thr - 1) / thr, thr, 0, stream>>>(
                gin_W1 + (size_t)c * H * H, bpGin1 + (size_t)c * H * H, H, H);
            pack_b_kernel<<<(n1 + thr - 1) / thr, thr, 0, stream>>>(
                gin_W2 + (size_t)c * H * H, bpGin2 + (size_t)c * H * H, H, H);
        }
        int n2 = 2 * H * H;
        pack_b_kernel<<<(n2 + thr - 1) / thr, thr, 0, stream>>>(out_W1, bpOutW1, 2 * H, H);
        int n3 = H * 128;
        pack_b_kernel<<<(n3 + thr - 1) / thr, thr, 0, stream>>>(out_W2, bpOutW2, H, 128);
    }

    // 2. time embedding + 1/std per graph, then per-edge lookups
    temb_kernel<<<G, H, 0, stream>>>(t, fw, temb_W, temb_b, d1W, d1b, tembG, stdinvG);
    e2g_kernel<<<(E + 255) / 256, 256, 0, stream>>>(src, batch, tembG, stdinvG,
                                                    tembE, stdinvE, E);

    // 3. bond_attr (fused input-MLP + temb + bond_emb gather)
    {
        dim3 grid((E + 127) / 128, H / 16);
        dembed_gemm_kernel<<<grid, 256, 0, stream>>>(elen, in_W1, in_b1, bpInW2, in_b2,
                                                     tembE, edge_type, bond_emb, bond, E);
    }

    // 4. h0 = atom embedding gather
    {
        long thr = (long)N * 64;
        gather_atoms_kernel<<<(unsigned)((thr + 255) / 256), 256, 0, stream>>>(
            node_type, atom_emb, hA, N);
    }

    // 5. GIN convolutions (ping-pong hA <-> hB)
    float* h_in = hA;
    float* h_out = hB;
    for (int c = 0; c < 4; ++c) {
        long n4 = (long)N * H / 4;
        zero_kernel<<<(unsigned)((n4 + 255) / 256), 256, 0, stream>>>(aggr, n4);
        long mthr = (long)E * 32;
        message_kernel<<<(unsigned)((mthr + 255) / 256), 256, 0, stream>>>(
            h_in, bond, src, dst, aggr, E);
        dim3 grid((N + 127) / 128, H / 16);
        gin_gemm1_kernel<<<grid, 256, 0, stream>>>(h_in, aggr,
                                                   bpGin1 + (size_t)c * H * H,
                                                   gin_b1 + (size_t)c * H, tmp, N);
        gin_gemm2_kernel<<<grid, 256, 0, stream>>>(tmp, bpGin2 + (size_t)c * H * H,
                                                   gin_b2 + (size_t)c * H,
                                                   h_in, h_out, N, c < 3 ? 1 : 0);
        float* sw = h_in; h_in = h_out; h_out = sw;
    }
    // after 4 convs, h_in holds the final node features

    // 6. output MLP over edges
    {
        dim3 grid1((E + 127) / 128, H / 16);
        out_gemm1_kernel<<<grid1, 256, 0, stream>>>(h_in, bond, src, dst, bpOutW1,
                                                    out_b1, y1, E);
        dim3 grid2((E + 127) / 128, 128 / 16);
        out_gemm2_kernel<<<grid2, 256, 0, stream>>>(y1, bpOutW2, out_b2, y2, E);
        long fthr = (long)E * 32;
        final_kernel<<<(unsigned)((fthr + 255) / 256), 256, 0, stream>>>(
            y2, out_W3, out_b3, stdinvE, scores, E);
    }
    (void)n_in; (void)out_size; (void)ws_size;
}
